// GPT_40123584479561
// MI455X (gfx1250) — matrix operand order repaired and checked by reading, not compiled, not run on hardware
//
#include <hip/hip_runtime.h>

// GPT-2-ish forward: B=2 T=2048 C=1024 V=50257 L=6 H=16 HD=64
constexpr int cB = 2, cT = 2048, cC = 1024, cV = 50257, cL = 6, cH = 16, cHD = 64;
constexpr int cBT = cB * cT;

typedef __attribute__((ext_vector_type(16))) __bf16 v16bf;
typedef __attribute__((ext_vector_type(8)))  float  v8f;
typedef __attribute__((ext_vector_type(4)))  unsigned int v4u;
typedef __attribute__((ext_vector_type(8)))  int    v8i;
typedef __attribute__((ext_vector_type(4)))  int    v4i;

__device__ inline v8f wmma_bf16(v16bf a, v16bf b, v8f c) {
    return __builtin_amdgcn_wmma_f32_16x16x32_bf16(false, a, false, b, (short)0, c, false, false);
}

// ---- Tensor Data Mover: DMA a 2D bf16 tile (tileD0 x tileD1) into LDS ----
// D# per CDNA5 ISA ch.8: group0 = {count, lds_addr, global_addr, type=2},
// group1 = {data_size=2B, tensor dims (rel. to tile start, for OOB zero-fill),
//           tile dims, dim0 stride}. Groups 2/3 zero (2D tensor).
__device__ inline void tdm_load_2d(unsigned ldsOff, const __bf16* gptr,
                                   int tensorD0, int tensorD1,
                                   int tileD0, int tileD1, int stride0) {
    unsigned long long ga = (unsigned long long)(size_t)gptr;
    v4u g0;
    g0[0] = 1u;                                               // count=1, user mode
    g0[1] = ldsOff;                                           // LDS byte address
    g0[2] = (unsigned)(ga & 0xFFFFFFFFu);                     // global_addr[31:0]
    g0[3] = (unsigned)((ga >> 32) & 0x1FFFFFFu) | (2u << 30); // addr[56:32] | type=2
    v8i g1;
    g1[0] = 1 << 16;                                          // data_size=1 (2 bytes)
    g1[1] = (int)(((unsigned)tensorD0 & 0xFFFFu) << 16);      // tensor_dim0[15:0]
    g1[2] = (int)(((unsigned)tensorD0 >> 16) |
                  (((unsigned)tensorD1 & 0xFFFFu) << 16));    // dim0[31:16] | dim1[15:0]
    g1[3] = (int)(((unsigned)tensorD1 >> 16) |
                  ((unsigned)tileD0 << 16));                  // dim1[31:16] | tile_dim0
    g1[4] = (int)((unsigned)tileD1 & 0xFFFFu);                // tile_dim1 (tile_dim2=0)
    g1[5] = stride0;                                          // tensor_dim0_stride[31:0]
    g1[6] = 0;
    g1[7] = 0;
    v4i z4 = {0, 0, 0, 0};
#if __clang_major__ >= 23
    v8i z8 = {0, 0, 0, 0, 0, 0, 0, 0};
    __builtin_amdgcn_tensor_load_to_lds(g0, g1, z4, z4, z8, 0);
#else
    __builtin_amdgcn_tensor_load_to_lds(g0, g1, z4, z4, 0);
#endif
}

// ---- fragment loaders (layouts per CDNA5 ISA 7.12.2, wave32) ----
// A (16x32 MxK, bf16, row-major src): lane<16 -> M=lane, K = k0+{0..7,16..23}
//                                     lane>=16 -> M=lane-16, K = k0+{8..15,24..31}
__device__ inline v16bf load_frag_a(const __bf16* A, int lda, int m0, int k0, int lane) {
    int row = m0 + (lane & 15);
    int kb  = k0 + ((lane >> 4) << 3);
    const __bf16* p = A + (size_t)row * lda + kb;
    union { uint4 u[2]; v16bf v; } u;
    u.u[0] = *reinterpret_cast<const uint4*>(p);
    u.u[1] = *reinterpret_cast<const uint4*>(p + 16);
    return u.v;
}

// B (32x16 KxN) from bf16 W[N,K] row-major (column n of B = row n of W):
// lane<16 -> N=lane, K=k0+{0..7,16..23} ; lane>=16 -> N=lane-16, K=k0+{8..15,24..31}
__device__ inline v16bf load_frag_b_bf16(const __bf16* W, int ldw, int n0, int k0, int lane) {
    int row = n0 + (lane & 15);
    const __bf16* p = W + (size_t)row * ldw + k0 + ((lane >> 4) << 3);
    union { uint4 u[2]; v16bf v; } u;
    u.u[0] = *reinterpret_cast<const uint4*>(p);
    u.u[1] = *reinterpret_cast<const uint4*>(p + 16);
    return u.v;
}

// ---- weight conversion f32 -> bf16 (once per launch; halves GEMM weight bytes) ----
__global__ __launch_bounds__(256) void cvt_bf16_kernel(const float* __restrict__ s,
                                                       __bf16* __restrict__ d, size_t n) {
    size_t i = (size_t)blockIdx.x * 256 + threadIdx.x;
    if (i < n) d[i] = (__bf16)s[i];
}

// ---- embedding: x = tok_emb[idx] + pos_emb ----
__global__ __launch_bounds__(256) void embed_kernel(const int* __restrict__ idx,
                                                    const float* __restrict__ tok,
                                                    const float* __restrict__ pos,
                                                    float* __restrict__ x) {
    size_t i = (size_t)blockIdx.x * blockDim.x + threadIdx.x;
    if (i >= (size_t)cBT * cC) return;
    int c = (int)(i % cC);
    size_t bt = i / cC;
    int t = (int)(bt % cT);
    x[i] = tok[(size_t)idx[bt] * cC + c] + pos[(size_t)t * cC + c];
}

// ---- rmsnorm: f32 row -> bf16 row (scaled) ----
__global__ __launch_bounds__(256) void rmsnorm_kernel(const float* __restrict__ x,
                                                      const float* __restrict__ scale,
                                                      __bf16* __restrict__ out) {
    int row = blockIdx.x;
    const float* xr = x + (size_t)row * cC;
    float ss = 0.f;
    for (int i = threadIdx.x; i < cC; i += 256) { float v = xr[i]; ss += v * v; }
    __shared__ float red[256];
    red[threadIdx.x] = ss; __syncthreads();
    for (int s = 128; s > 0; s >>= 1) {
        if ((int)threadIdx.x < s) red[threadIdx.x] += red[threadIdx.x + s];
        __syncthreads();
    }
    float inv = rsqrtf(red[0] / (float)cC + 1e-6f);
    for (int i = threadIdx.x; i < cC; i += 256)
        out[(size_t)row * cC + i] = (__bf16)(xr[i] * inv * scale[i]);
}

// ---- WMMA GEMM with TDM double-buffered LDS staging ----
// Out[M,N] = A[M,K](bf16) * W[N,K](bf16)^T.
// 256 threads = 8 waves; block tile 64(M) x 256(N); wave tile 32x64
// (2 A-frags x 4 B-frags = 8 WMMAs per 32-wide K step -> 2x flops per LDS byte).
// Wave 0 drives the Tensor Data Mover: per K step it DMAs the A-tile
// (64x32, 4KB) and B-tile (256x32, 16KB) into LDS, double buffered,
// synchronized with s_wait_tensorcnt + workgroup barrier. TDM zero-fills
// out-of-range rows, which handles the ragged head matrix (N=50257).
__global__ __launch_bounds__(256) void gemm_kernel(const __bf16* __restrict__ A, int lda,
                                                   const __bf16* __restrict__ W, int ldw,
                                                   float* __restrict__ outF,
                                                   __bf16* __restrict__ outB,
                                                   const float* __restrict__ resid,
                                                   int M, int N, int K, int act) {
    __shared__ __bf16 sA[2][64 * 32];
    __shared__ __bf16 sB[2][256 * 32];
    int lane = threadIdx.x & 31;
    int wave = threadIdx.x >> 5;
    int wm = (wave & 1) * 32;      // M offset of this wave inside the block tile
    int wn = (wave >> 1) * 64;     // N offset of this wave inside the block tile
    int m0 = blockIdx.y * 64;
    int n0 = blockIdx.x * 256;
    int nsteps = K / 32;

    const __bf16* Abase = A + (size_t)m0 * lda;
    const __bf16* Wbase = W + (size_t)n0 * ldw;

    if (wave == 0) {   // whole-wave guard: tensor ops ignore EXEC
        tdm_load_2d((unsigned)(size_t)&sA[0][0], Abase, K, M - m0, 32, 64, lda);
        tdm_load_2d((unsigned)(size_t)&sB[0][0], Wbase, K, N - n0, 32, 256, ldw);
    }

    v8f acc[2][4] = {};
    for (int s = 0; s < nsteps; ++s) {
        int buf = s & 1;
        if (wave == 0) {
            if (s + 1 < nsteps) {
                int k1 = (s + 1) * 32;
                tdm_load_2d((unsigned)(size_t)&sA[buf ^ 1][0], Abase + k1,
                            K - k1, M - m0, 32, 64, lda);
                tdm_load_2d((unsigned)(size_t)&sB[buf ^ 1][0], Wbase + k1,
                            K - k1, N - n0, 32, 256, ldw);
                __builtin_amdgcn_s_wait_tensorcnt(2);  // stage s done; s+1 in flight
            } else {
                __builtin_amdgcn_s_wait_tensorcnt(0);
            }
        }
        __syncthreads();
        const __bf16* tA = &sA[buf][0];
        const __bf16* tB = &sB[buf][0];
        v16bf a0 = load_frag_a(tA, 32, wm, 0, lane);
        v16bf a1 = load_frag_a(tA, 32, wm + 16, 0, lane);
#pragma unroll
        for (int t = 0; t < 4; ++t) {
            v16bf b = load_frag_b_bf16(tB, 32, wn + t * 16, 0, lane);
            acc[0][t] = wmma_bf16(a0, b, acc[0][t]);
            acc[1][t] = wmma_bf16(a1, b, acc[1][t]);
        }
        __syncthreads();   // all reads done before stage s+2 overwrites this buffer
    }

    int ncol = lane & 15;
#pragma unroll
    for (int i = 0; i < 2; ++i) {
        int mrow = m0 + wm + i * 16 + (lane >> 4) * 8;
#pragma unroll
        for (int t = 0; t < 4; ++t) {
            int n = n0 + wn + t * 16 + ncol;
            if (n >= N) continue;
#pragma unroll
            for (int r = 0; r < 8; ++r) {
                size_t off = (size_t)(mrow + r) * N + n;
                float v = acc[i][t][r];
                if (resid) v += resid[off];
                if (act) { float rl = v > 0.f ? v : 0.f; v = rl * rl; }
                if (outF) outF[off] = v;
                if (outB) outB[off] = (__bf16)v;
            }
        }
    }
}

// ---- V transpose: vt[b,h,d,t] = qkv[b*T+t, 2C + h*HD + d] ----
__global__ __launch_bounds__(256) void transpose_v_kernel(const __bf16* __restrict__ qkv,
                                                          __bf16* __restrict__ vt) {
    size_t i = (size_t)blockIdx.x * blockDim.x + threadIdx.x;
    if (i >= (size_t)cBT * cC) return;
    int t = (int)(i % cT);
    size_t rest = i / cT;
    int d = (int)(rest % cHD); rest /= cHD;
    int h = (int)(rest % cH);
    int b = (int)(rest / cH);
    vt[i] = qkv[(size_t)(b * cT + t) * (3 * cC) + 2 * cC + h * cHD + d];
}

// ---- flash attention: one wave per (b, h, 16-query tile) ----
__global__ __launch_bounds__(32) void attn_kernel(const __bf16* __restrict__ qkv,
                                                  const __bf16* __restrict__ vt,
                                                  __bf16* __restrict__ y) {
    __shared__ __bf16 pl[16 * 32];           // P tile staged through LDS: C/D -> A layout
    int lane = threadIdx.x & 31;
    int qt = blockIdx.x & (cT / 16 - 1);
    int h  = (blockIdx.x / (cT / 16)) & (cH - 1);
    int b  = blockIdx.x / ((cT / 16) * cH);
    int q0 = qt * 16;

    const __bf16* Q  = qkv + (size_t)b * cT * 3 * cC + h * cHD;
    const __bf16* Kp = qkv + (size_t)b * cT * 3 * cC + cC + h * cHD;
    const __bf16* Vt = vt + (size_t)(b * cH + h) * cHD * cT;

    v16bf aq0 = load_frag_a(Q, 3 * cC, q0, 0, lane);
    v16bf aq1 = load_frag_a(Q, 3 * cC, q0, 32, lane);

    v8f o[4] = {};
    float mrow[8], lrow[8];
#pragma unroll
    for (int r = 0; r < 8; ++r) { mrow[r] = -1e30f; lrow[r] = 0.f; }
    const float scale = 0.125f;              // 1/sqrt(64)
    int half8 = (lane >> 4) * 8;
    int ncol = lane & 15;

    for (int t0 = 0; t0 <= q0 + 15; t0 += 32) {
        v8f sa = {}, sb = {};
        sa = wmma_bf16(aq0, load_frag_b_bf16(Kp, 3 * cC, t0,      0,  lane), sa);
        sa = wmma_bf16(aq1, load_frag_b_bf16(Kp, 3 * cC, t0,      32, lane), sa);
        sb = wmma_bf16(aq0, load_frag_b_bf16(Kp, 3 * cC, t0 + 16, 0,  lane), sb);
        sb = wmma_bf16(aq1, load_frag_b_bf16(Kp, 3 * cC, t0 + 16, 32, lane), sb);

        float pa[8], pb[8], fsc[8];
#pragma unroll
        for (int r = 0; r < 8; ++r) {
            int qrow = q0 + half8 + r;
            float va = sa[r] * scale;
            float vb = sb[r] * scale;
            if (t0 + ncol > qrow)      va = -1e30f;   // causal mask
            if (t0 + 16 + ncol > qrow) vb = -1e30f;
            float mx = fmaxf(va, vb);
            for (int s = 1; s < 16; s <<= 1) mx = fmaxf(mx, __shfl_xor(mx, s, 32));
            mx = fmaxf(mx, mrow[r]);
            float f  = __expf(mrow[r] - mx);
            float ea = __expf(va - mx);
            float eb = __expf(vb - mx);
            float sm = ea + eb;
            for (int s = 1; s < 16; s <<= 1) sm += __shfl_xor(sm, s, 32);
            lrow[r] = lrow[r] * f + sm;
            mrow[r] = mx;
            pa[r] = ea; pb[r] = eb; fsc[r] = f;
        }
#pragma unroll
        for (int t = 0; t < 4; ++t)
#pragma unroll
            for (int r = 0; r < 8; ++r) o[t][r] *= fsc[r];

#pragma unroll
        for (int r = 0; r < 8; ++r) {
            int row = half8 + r;
            pl[row * 32 + ncol]      = (__bf16)pa[r];
            pl[row * 32 + 16 + ncol] = (__bf16)pb[r];
        }
        __syncthreads();
        v16bf pf = load_frag_a(pl, 32, 0, 0, lane);
        __syncthreads();

#pragma unroll
        for (int t = 0; t < 4; ++t)
            o[t] = wmma_bf16(pf, load_frag_b_bf16(Vt, cT, t * 16, t0, lane), o[t]);
    }

#pragma unroll
    for (int t = 0; t < 4; ++t)
#pragma unroll
        for (int r = 0; r < 8; ++r) {
            int m = half8 + r;
            int col = h * cHD + t * 16 + ncol;
            y[(size_t)(b * cT + q0 + m) * cC + col] = (__bf16)(o[t][r] / lrow[r]);
        }
}

// ---- loss: mean NLL over rows of logits ----
__global__ void zero_loss_kernel(float* loss) { *loss = 0.f; }

__global__ __launch_bounds__(256) void loss_kernel(const float* __restrict__ logits,
                                                   const int* __restrict__ targets,
                                                   float* __restrict__ loss) {
    int row = blockIdx.x;
    const float* lr = logits + (size_t)row * cV;
    __shared__ float red[256];
    float mx = -1e30f;
    for (int i = threadIdx.x; i < cV; i += 256) mx = fmaxf(mx, lr[i]);
    red[threadIdx.x] = mx; __syncthreads();
    for (int s = 128; s > 0; s >>= 1) {
        if ((int)threadIdx.x < s) red[threadIdx.x] = fmaxf(red[threadIdx.x], red[threadIdx.x + s]);
        __syncthreads();
    }
    mx = red[0]; __syncthreads();
    float se = 0.f;
    for (int i = threadIdx.x; i < cV; i += 256) se += __expf(lr[i] - mx);
    red[threadIdx.x] = se; __syncthreads();
    for (int s = 128; s > 0; s >>= 1) {
        if ((int)threadIdx.x < s) red[threadIdx.x] += red[threadIdx.x + s];
        __syncthreads();
    }
    if (threadIdx.x == 0) {
        float lse = mx + __logf(red[0]);
        float nll = lse - lr[targets[row]];
        atomicAdd(loss, nll * (1.0f / (float)cBT));
    }
}

static inline int gridFor(size_t n) { return (int)((n + 255) / 256); }

extern "C" void kernel_launch(void* const* d_in, const int* in_sizes, int n_in,
                              void* d_out, int out_size, void* d_ws, size_t ws_size,
                              hipStream_t stream) {
    (void)in_sizes; (void)n_in; (void)out_size; (void)ws_size;
    const int*   idx        = (const int*)d_in[0];
    const int*   targets    = (const int*)d_in[1];
    const float* tok_emb    = (const float*)d_in[2];
    const float* pos_emb    = (const float*)d_in[3];
    const float* n1_scale   = (const float*)d_in[4];   // [L,C]
    const float* qkv_w      = (const float*)d_in[5];   // [L,3C,C]
    const float* proj_w     = (const float*)d_in[6];   // [L,C,C]
    const float* n2_scale   = (const float*)d_in[7];   // [L,C]
    const float* fc1_w      = (const float*)d_in[8];   // [L,4C,C]
    const float* fc2_w      = (const float*)d_in[9];   // [L,C,4C]
    const float* norm_scale = (const float*)d_in[10];  // [C]
    const float* head_w     = (const float*)d_in[11];  // [V,C]

    float* logits = (float*)d_out;                     // [B*T, V]
    float* loss   = logits + (size_t)cBT * cV;         // scalar

    // workspace layout
    char* w = (char*)d_ws;
    float*  x    = (float*)w;  w += (size_t)cBT * cC * 4;
    __bf16* h    = (__bf16*)w; w += (size_t)cBT * cC * 2;
    __bf16* qkvb = (__bf16*)w; w += (size_t)cBT * 3 * cC * 2;
    __bf16* vt   = (__bf16*)w; w += (size_t)cBT * cC * 2;
    __bf16* y    = (__bf16*)w; w += (size_t)cBT * cC * 2;
    __bf16* mlp  = (__bf16*)w; w += (size_t)cBT * 4 * cC * 2;
    __bf16* xn   = (__bf16*)w; w += (size_t)cBT * cC * 2;
    const size_t nQ = (size_t)cL * 3 * cC * cC;
    const size_t nP = (size_t)cL * cC * cC;
    const size_t nF1 = (size_t)cL * 4 * cC * cC;
    const size_t nF2 = (size_t)cL * cC * 4 * cC;
    const size_t nHd = (size_t)cV * cC;
    __bf16* wqkv  = (__bf16*)w; w += nQ * 2;
    __bf16* wproj = (__bf16*)w; w += nP * 2;
    __bf16* wfc1  = (__bf16*)w; w += nF1 * 2;
    __bf16* wfc2  = (__bf16*)w; w += nF2 * 2;
    __bf16* whead = (__bf16*)w; w += nHd * 2;

    // one-time (per launch) weight conversion to bf16
    cvt_bf16_kernel<<<gridFor(nQ),  256, 0, stream>>>(qkv_w,  wqkv,  nQ);
    cvt_bf16_kernel<<<gridFor(nP),  256, 0, stream>>>(proj_w, wproj, nP);
    cvt_bf16_kernel<<<gridFor(nF1), 256, 0, stream>>>(fc1_w,  wfc1,  nF1);
    cvt_bf16_kernel<<<gridFor(nF2), 256, 0, stream>>>(fc2_w,  wfc2,  nF2);
    cvt_bf16_kernel<<<gridFor(nHd), 256, 0, stream>>>(head_w, whead, nHd);

    const int elemBlocks = gridFor((size_t)cBT * cC);
    embed_kernel<<<elemBlocks, 256, 0, stream>>>(idx, tok_emb, pos_emb, x);

    for (int l = 0; l < cL; ++l) {
        // attention block
        rmsnorm_kernel<<<cBT, 256, 0, stream>>>(x, n1_scale + (size_t)l * cC, h);
        gemm_kernel<<<dim3(3 * cC / 256, cBT / 64), 256, 0, stream>>>(
            h, cC, wqkv + (size_t)l * 3 * cC * cC, cC,
            nullptr, qkvb, nullptr, cBT, 3 * cC, cC, 0);
        transpose_v_kernel<<<elemBlocks, 256, 0, stream>>>(qkvb, vt);
        attn_kernel<<<cB * cH * (cT / 16), 32, 0, stream>>>(qkvb, vt, y);
        gemm_kernel<<<dim3(cC / 256, cBT / 64), 256, 0, stream>>>(
            y, cC, wproj + (size_t)l * cC * cC, cC,
            x, nullptr, x, cBT, cC, cC, 0);
        // MLP block (squared ReLU)
        rmsnorm_kernel<<<cBT, 256, 0, stream>>>(x, n2_scale + (size_t)l * cC, h);
        gemm_kernel<<<dim3(4 * cC / 256, cBT / 64), 256, 0, stream>>>(
            h, cC, wfc1 + (size_t)l * 4 * cC * cC, cC,
            nullptr, mlp, nullptr, cBT, 4 * cC, cC, 1);
        gemm_kernel<<<dim3(cC / 256, cBT / 64), 256, 0, stream>>>(
            mlp, 4 * cC, wfc2 + (size_t)l * cC * 4 * cC, 4 * cC,
            x, nullptr, x, cBT, cC, 4 * cC, 0);
    }

    // final norm + head + loss
    rmsnorm_kernel<<<cBT, 256, 0, stream>>>(x, norm_scale, xn);
    gemm_kernel<<<dim3((cV + 255) / 256, cBT / 64), 256, 0, stream>>>(
        xn, cC, whead, cC, logits, nullptr, nullptr, cBT, cV, cC, 0);
    zero_loss_kernel<<<1, 1, 0, stream>>>(loss);
    loss_kernel<<<cBT, 256, 0, stream>>>(logits, targets, loss);
}
